// _TransformerEncoderBlock_38989713113794
// MI455X (gfx1250) — matrix operand order repaired and checked by reading, not compiled, not run on hardware
//
#include <hip/hip_runtime.h>
#include <hip/hip_bf16.h>
#include <math.h>

#define B_  4
#define N_  2048
#define D_  1024
#define H_  16
#define HD_ 64
#define DF_ 4096
#define M_  (B_ * N_)

#define GP 72  // padded LDS row stride for 64-wide tiles (bank-conflict-free)

typedef __bf16 bf16_t;
typedef __attribute__((ext_vector_type(16))) __bf16 bf16x16;
typedef __attribute__((ext_vector_type(8)))  float  floatx8;
typedef __attribute__((ext_vector_type(4)))  int    i4v;

// POD 16-byte vector so it can live in a union (HIP's uint4 has ctors).
struct alignas(16) U4 { unsigned int x, y, z, w; };

union FragBF { U4 u[2]; bf16x16 v; };

// ---------------------------------------------------------------------------
// CDNA5 async global->LDS copies (ASYNCcnt path), guarded so we always compile.
// Probe-confirmed signature: (int4 AS1* src, int4 AS3* dst, imm offset, imm cpol)
// ---------------------------------------------------------------------------
#if __has_builtin(__builtin_amdgcn_global_load_async_to_lds_b128) && \
    __has_builtin(__builtin_amdgcn_s_wait_asynccnt)
#define HAS_ASYNC_LDS 1
#else
#define HAS_ASYNC_LDS 0
#endif

__device__ __forceinline__ void async_b128(void* lds, const void* g) {
#if HAS_ASYNC_LDS
  __builtin_amdgcn_global_load_async_to_lds_b128(
      (__attribute__((address_space(1))) i4v*)(g),
      (__attribute__((address_space(3))) i4v*)(lds),
      0, 0);
#else
  *(U4*)lds = *(const U4*)g;
#endif
}

__device__ __forceinline__ void async_wait0() {
#if HAS_ASYNC_LDS
  __builtin_amdgcn_s_wait_asynccnt(0);
#endif
}

__device__ __forceinline__ floatx8 wmma_bf16(const FragBF& a, const FragBF& b, floatx8 c) {
  // D = A(16x32 bf16) * B(32x16 bf16) + C(16x16 f32)
  return __builtin_amdgcn_wmma_f32_16x16x32_bf16(false, a.v, false, b.v, (short)0, c,
                                                 false, false);
}

__device__ __forceinline__ float gelu_exact(float x) {
  return 0.5f * x * (1.0f + erff(x * 0.70710678118654752f));
}

// ---------------------------------------------------------------------------
// f32 -> bf16 elementwise
// ---------------------------------------------------------------------------
__global__ __launch_bounds__(256) void cvt_bf16_kernel(const float* __restrict__ x,
                                                       bf16_t* __restrict__ o, int n) {
  int i = blockIdx.x * 256 + threadIdx.x;
  if (i < n) o[i] = (bf16_t)x[i];
}

// ---------------------------------------------------------------------------
// W (K x N, f32, row-major) -> Wt (N x K, bf16)  : makes GEMM B-frags K-contiguous
// ---------------------------------------------------------------------------
__global__ __launch_bounds__(256) void transpose_cvt_kernel(const float* __restrict__ W,
                                                            bf16_t* __restrict__ Wt,
                                                            int K, int N) {
  __shared__ float tile[32][33];
  const int tid = threadIdx.x;
  const int tx = tid & 31, ty = tid >> 5;  // 32 x 8
  const int nb = blockIdx.x * 32, kb = blockIdx.y * 32;
#pragma unroll
  for (int i = 0; i < 4; ++i) {
    int k = kb + ty + i * 8;
    tile[ty + i * 8][tx] = W[(size_t)k * N + nb + tx];
  }
  __syncthreads();
#pragma unroll
  for (int i = 0; i < 4; ++i) {
    int n = nb + ty + i * 8;
    Wt[(size_t)n * K + kb + tx] = (bf16_t)tile[tx][ty + i * 8];
  }
}

// ---------------------------------------------------------------------------
// V [B, N, H*HD] (bf16) -> VT [B, H, HD, N] (bf16): hoists the per-head
// transpose out of the attention hot loop so V chunks stage via async b128.
// ---------------------------------------------------------------------------
__global__ __launch_bounds__(256) void transpose_v_kernel(const bf16_t* __restrict__ V,
                                                          bf16_t* __restrict__ VT) {
  __shared__ bf16_t tile[32][33];
  const int tid = threadIdx.x;
  const int tx = tid & 31, ty = tid >> 5;  // 32 x 8
  const int t0 = blockIdx.x * 32;          // token base
  const int h  = blockIdx.y >> 1;          // 2 hd-tiles per head (HD=64)
  const int d0 = (blockIdx.y & 1) * 32;    // hd base within head
  const int bb = blockIdx.z;
  const bf16_t* src = V + (size_t)bb * N_ * D_ + h * HD_ + d0;
#pragma unroll
  for (int i = 0; i < 4; ++i) {
    int t = ty + i * 8;
    tile[t][tx] = src[(size_t)(t0 + t) * D_ + tx];  // [token][hd]
  }
  __syncthreads();
  bf16_t* dst = VT + (((size_t)bb * H_ + h) * HD_ + d0) * N_ + t0;
#pragma unroll
  for (int i = 0; i < 4; ++i) {
    int dd = ty + i * 8;
    dst[(size_t)dd * N_ + tx] = tile[tx][dd];       // [hd][token]
  }
}

// ---------------------------------------------------------------------------
// WMMA GEMM:  C[M,N] = A[M,K](bf16) * Bt[N,K](bf16)^T + bias
// MODE 0: store bf16    MODE 1: store bf16 after exact GELU    MODE 2: store f32
// Block: 128 thr = 4 waves (2x2), block tile 128x128, wave tile 64x64, BK=64.
// Double-buffered padded LDS + async global->LDS: 32 WMMAs per barrier.
// ---------------------------------------------------------------------------
template <int MODE>
__global__ __launch_bounds__(128) void gemm_bf16_kernel(const bf16_t* __restrict__ A,
                                                        const bf16_t* __restrict__ Bt,
                                                        const float* __restrict__ bias,
                                                        void* __restrict__ Cout,
                                                        int M, int N, int K) {
  __shared__ bf16_t sA[2][128 * GP];
  __shared__ bf16_t sB[2][128 * GP];
  const int tid  = threadIdx.x;
  const int lane = tid & 31;
  const int wave = tid >> 5;
  const int ll = lane & 15, lh = lane >> 4;
  const int bm = blockIdx.x * 128;
  const int bn = blockIdx.y * 128;
  const int wm = (wave & 1) * 64;   // 2 waves down M
  const int wn = (wave >> 1) * 64;  // 2 waves across N

  floatx8 acc[4][4] = {};

  // Stage one 128x64 A tile + 128x64 Bt tile (16 async b128 per thread).
  // Chunk id s*128+tid: 8 consecutive lanes cover one 128B row contiguously.
  auto stage = [&](int buf, int k0) {
#pragma unroll
    for (int s = 0; s < 8; ++s) {
      int e = (s * 128 + tid) * 8;
      int row = e >> 6, col = e & 63;
      async_b128(&sA[buf][row * GP + col], &A[(size_t)(bm + row) * K + k0 + col]);
      async_b128(&sB[buf][row * GP + col], &Bt[(size_t)(bn + row) * K + k0 + col]);
    }
  };

  stage(0, 0);
  int buf = 0;
  for (int k0 = 0; k0 < K; k0 += 64) {
    async_wait0();    // this wave's copies for `buf` have landed in LDS
    __syncthreads();  // everyone's copies landed; prior reads of buf^1 done
    if (k0 + 64 < K) stage(buf ^ 1, k0 + 64);  // overlap next copy with compute

#pragma unroll
    for (int kh = 0; kh < 2; ++kh) {
      // A-frag (16x32): lane ll = row; halves -> k = lh*16 + 0..15 (contiguous)
      FragBF af[4], bfr[4];
#pragma unroll
      for (int mi = 0; mi < 4; ++mi) {
        int r = wm + mi * 16 + ll;
        af[mi].u[0] = *(const U4*)&sA[buf][r * GP + kh * 32 + lh * 16];
        af[mi].u[1] = *(const U4*)&sA[buf][r * GP + kh * 32 + lh * 16 + 8];
      }
      // B-frag (32x16): lane ll = col n; halves -> k = lh*16 + 0..15 (contiguous)
#pragma unroll
      for (int ni = 0; ni < 4; ++ni) {
        int r = wn + ni * 16 + ll;
        const U4* p = (const U4*)&sB[buf][r * GP + kh * 32 + lh * 16];
        bfr[ni].u[0] = p[0];
        bfr[ni].u[1] = p[1];
      }
#pragma unroll
      for (int mi = 0; mi < 4; ++mi)
#pragma unroll
        for (int ni = 0; ni < 4; ++ni) acc[mi][ni] = wmma_bf16(af[mi], bfr[ni], acc[mi][ni]);
    }
    buf ^= 1;
  }

  // Epilogue. C layout: lane ll = col, row = lh*8 + r (coalesced across 16 lanes).
#pragma unroll
  for (int mi = 0; mi < 4; ++mi) {
#pragma unroll
    for (int ni = 0; ni < 4; ++ni) {
      int n = bn + wn + ni * 16 + ll;
      float bv = bias[n];
#pragma unroll
      for (int r = 0; r < 8; ++r) {
        int m = bm + wm + mi * 16 + lh * 8 + r;
        float v = acc[mi][ni][r] + bv;
        if (MODE == 1) v = gelu_exact(v);
        if (MODE == 2)
          ((float*)Cout)[(size_t)m * N + n] = v;
        else
          ((bf16_t*)Cout)[(size_t)m * N + n] = (bf16_t)v;
      }
    }
  }
}

// ---------------------------------------------------------------------------
// Flash attention: grid (N/64, H, B), 128 thr = 4 waves, each wave 16 q-rows.
// S = (Q K^T)/sqrt(D) streamed over 64-key chunks with online softmax.
// K and VT chunks double-buffered, staged with pure async b128 copies.
// Row sums computed with WMMA (P * ones) instead of cross-lane shuffles.
// ---------------------------------------------------------------------------
__global__ __launch_bounds__(128) void attention_kernel(const bf16_t* __restrict__ Q,
                                                        const bf16_t* __restrict__ Kp,
                                                        const bf16_t* __restrict__ VT,
                                                        bf16_t* __restrict__ Out) {
  __shared__ bf16_t sK[2][64 * GP];   // [key][hd]
  __shared__ bf16_t sVT[2][64 * GP];  // [hd][key]
  __shared__ bf16_t sP[4][16 * GP];   // per-wave P, [qrow][key]

  const int tid  = threadIdx.x;
  const int lane = tid & 31;
  const int wave = tid >> 5;
  const int ll = lane & 15, lh = lane >> 4;
  const int q0 = blockIdx.x * 64;
  const int h  = blockIdx.y;
  const int bb = blockIdx.z;

  // Q fragments for this wave's 16 rows (HD=64 -> 2 frags of 16x32), kept in regs.
  const bf16_t* qp = Q + ((size_t)bb * N_ + (q0 + wave * 16 + ll)) * D_ + h * HD_;
  FragBF qf[2];
#pragma unroll
  for (int f = 0; f < 2; ++f) {
    qf[f].u[0] = *(const U4*)&qp[f * 32 + lh * 16];
    qf[f].u[1] = *(const U4*)&qp[f * 32 + lh * 16 + 8];
  }

  // Constant B-matrix of bf16 1.0s (0x3F80) for WMMA row-sum of P.
  FragBF ones;
  ones.u[0] = U4{0x3F803F80u, 0x3F803F80u, 0x3F803F80u, 0x3F803F80u};
  ones.u[1] = ones.u[0];

  floatx8 o[4] = {};
  floatx8 lacc = {};
  float rm[8];
#pragma unroll
  for (int r = 0; r < 8; ++r) rm[r] = -1e30f;

  const bf16_t* kbase  = Kp + (size_t)bb * N_ * D_ + h * HD_;
  const bf16_t* vtbase = VT + ((size_t)bb * H_ + h) * HD_ * N_;

  // Stage K rows 0..31 (32x64) + VT rows 0..31 (32x64) of the chunk, 4 async b128/thread.
  auto stage = [&](int sbuf, int j0) {
#pragma unroll
    for (int s = 0; s < 2; ++s) {
      int e = (s * 128 + tid) * 8;
      int r = e >> 6, c2 = e & 63;
      async_b128(&sK[sbuf][r * GP + c2], &kbase[(size_t)(j0 + r) * D_ + c2]);
      async_b128(&sVT[sbuf][r * GP + c2], &vtbase[(size_t)r * N_ + j0 + c2]);
    }
  };

  stage(0, 0);
  int buf = 0;
  for (int j0 = 0; j0 < N_; j0 += 64) {
    async_wait0();    // chunk `buf` landed
    __syncthreads();  // all waves' copies landed; prior reads of buf^1 done
    if (j0 + 64 < N_) stage(buf ^ 1, j0 + 64);  // prefetch next chunk

    // S tiles: 4 x (16q x 16k), K-dim = HD = 64 -> 2 WMMAs each.
    floatx8 st[4];
#pragma unroll
    for (int kk = 0; kk < 4; ++kk) {
      FragBF kf0, kf1;  // B-frag: lane ll = key col, halves = contiguous hd
      const U4* p0 = (const U4*)&sK[buf][(kk * 16 + ll) * GP + lh * 16];
      kf0.u[0] = p0[0]; kf0.u[1] = p0[1];
      const U4* p1 = (const U4*)&sK[buf][(kk * 16 + ll) * GP + 32 + lh * 16];
      kf1.u[0] = p1[0]; kf1.u[1] = p1[1];
      floatx8 s = {0, 0, 0, 0, 0, 0, 0, 0};
      s = wmma_bf16(qf[0], kf0, s);
      s = wmma_bf16(qf[1], kf1, s);
#pragma unroll
      for (int r = 0; r < 8; ++r) s[r] *= 0.03125f;  // 1/sqrt(D), D=1024
      st[kk] = s;
    }

    // Online softmax max: rows live across 16-lane groups, shfl_xor reduce.
    float nm[8], corr[8];
#pragma unroll
    for (int r = 0; r < 8; ++r) {
      float cm = fmaxf(fmaxf(st[0][r], st[1][r]), fmaxf(st[2][r], st[3][r]));
#pragma unroll
      for (int off = 1; off < 16; off <<= 1) cm = fmaxf(cm, __shfl_xor(cm, off, 32));
      float n2 = fmaxf(rm[r], cm);
      nm[r] = n2;
      corr[r] = __expf(rm[r] - n2);
      rm[r] = n2;
    }

    // P = exp(S - rowmax) -> LDS (C-layout to A-layout transpose via LDS).
#pragma unroll
    for (int kk = 0; kk < 4; ++kk)
#pragma unroll
      for (int r = 0; r < 8; ++r)
        sP[wave][(lh * 8 + r) * GP + kk * 16 + ll] = (bf16_t)__expf(st[kk][r] - nm[r]);

    // Rescale running O and l by the max correction.
#pragma unroll
    for (int r = 0; r < 8; ++r) {
      lacc[r] *= corr[r];
#pragma unroll
      for (int c = 0; c < 4; ++c) o[c][r] *= corr[r];
    }

    __syncthreads();  // sP visible to all lanes

    // O += P * V ; l += P * ones (WMMA row-sum, replicated over columns).
    FragBF pf[2];
#pragma unroll
    for (int f = 0; f < 2; ++f) {
      pf[f].u[0] = *(const U4*)&sP[wave][ll * GP + f * 32 + lh * 16];
      pf[f].u[1] = *(const U4*)&sP[wave][ll * GP + f * 32 + lh * 16 + 8];
    }
    lacc = wmma_bf16(pf[0], ones, lacc);
    lacc = wmma_bf16(pf[1], ones, lacc);
#pragma unroll
    for (int c = 0; c < 4; ++c) {
      FragBF vf0, vf1;
      const U4* p0 = (const U4*)&sVT[buf][(c * 16 + ll) * GP + lh * 16];
      vf0.u[0] = p0[0]; vf0.u[1] = p0[1];
      const U4* p1 = (const U4*)&sVT[buf][(c * 16 + ll) * GP + 32 + lh * 16];
      vf1.u[0] = p1[0]; vf1.u[1] = p1[1];
      o[c] = wmma_bf16(pf[0], vf0, o[c]);
      o[c] = wmma_bf16(pf[1], vf1, o[c]);
    }
    buf ^= 1;
  }

  // Normalize and write out (heads concatenated: [b, q, h*64 + hd]).
  bf16_t* obase = Out + ((size_t)bb * N_ + q0 + wave * 16) * D_ + h * HD_;
#pragma unroll
  for (int c = 0; c < 4; ++c) {
#pragma unroll
    for (int r = 0; r < 8; ++r) {
      int m = lh * 8 + r;
      obase[(size_t)m * D_ + c * 16 + ll] = (bf16_t)(o[c][r] / lacc[r]);
    }
  }
}

// ---------------------------------------------------------------------------
// y = LayerNorm(a + res) * g + b ; optionally also emits bf16 copy of y.
// One 256-thread block per row (D=1024 -> 4 elems/thread).
// ---------------------------------------------------------------------------
__global__ __launch_bounds__(256) void resln_kernel(const float* __restrict__ a,
                                                    const float* __restrict__ res,
                                                    const float* __restrict__ g,
                                                    const float* __restrict__ bta,
                                                    float* __restrict__ y,
                                                    bf16_t* __restrict__ yb) {
  const int row = blockIdx.x;
  const int tid = threadIdx.x;
  const int lane = tid & 31, wave = tid >> 5;
  __shared__ float red[8], red2[8];
  const size_t base = (size_t)row * D_;

  float v[4];
  float s = 0.f, sq = 0.f;
#pragma unroll
  for (int i = 0; i < 4; ++i) {
    int c = tid + i * 256;
    float t = a[base + c] + res[base + c];
    v[i] = t;
    s += t;
    sq += t * t;
  }
#pragma unroll
  for (int off = 16; off; off >>= 1) {
    s += __shfl_xor(s, off, 32);
    sq += __shfl_xor(sq, off, 32);
  }
  if (lane == 0) { red[wave] = s; red2[wave] = sq; }
  __syncthreads();
  if (tid == 0) {
    float S = 0.f, Q = 0.f;
#pragma unroll
    for (int i = 0; i < 8; ++i) { S += red[i]; Q += red2[i]; }
    red[0] = S;
    red2[0] = Q;
  }
  __syncthreads();
  const float mean = red[0] * (1.f / D_);
  const float var  = red2[0] * (1.f / D_) - mean * mean;
  const float inv  = rsqrtf(var + 1e-5f);
#pragma unroll
  for (int i = 0; i < 4; ++i) {
    int c = tid + i * 256;
    float o = (v[i] - mean) * inv * g[c] + bta[c];
    y[base + c] = o;
    if (yb) yb[base + c] = (bf16_t)o;
  }
}

// ---------------------------------------------------------------------------
extern "C" void kernel_launch(void* const* d_in, const int* in_sizes, int n_in,
                              void* d_out, int out_size, void* d_ws, size_t ws_size,
                              hipStream_t stream) {
  (void)in_sizes; (void)n_in; (void)out_size; (void)ws_size;
  const float* x   = (const float*)d_in[0];
  const float* Wq  = (const float*)d_in[1];
  const float* bq  = (const float*)d_in[2];
  const float* Wk  = (const float*)d_in[3];
  const float* bk  = (const float*)d_in[4];
  const float* Wv  = (const float*)d_in[5];
  const float* bv  = (const float*)d_in[6];
  const float* Wo  = (const float*)d_in[7];
  const float* bo  = (const float*)d_in[8];
  const float* g1  = (const float*)d_in[9];
  const float* be1 = (const float*)d_in[10];
  const float* W1  = (const float*)d_in[11];
  const float* b1  = (const float*)d_in[12];
  const float* W2  = (const float*)d_in[13];
  const float* b2  = (const float*)d_in[14];
  const float* g2  = (const float*)d_in[15];
  const float* be2 = (const float*)d_in[16];
  float* out = (float*)d_out;

  char* wp = (char*)d_ws;
  auto alloc = [&](size_t bytes) -> void* {
    void* p = (void*)wp;
    wp += (bytes + 255) & ~(size_t)255;
    return p;
  };
  bf16_t* xb  = (bf16_t*)alloc((size_t)M_ * D_ * 2);
  bf16_t* WqT = (bf16_t*)alloc((size_t)D_ * D_ * 2);
  bf16_t* WkT = (bf16_t*)alloc((size_t)D_ * D_ * 2);
  bf16_t* WvT = (bf16_t*)alloc((size_t)D_ * D_ * 2);
  bf16_t* WoT = (bf16_t*)alloc((size_t)D_ * D_ * 2);
  bf16_t* W1T = (bf16_t*)alloc((size_t)DF_ * D_ * 2);
  bf16_t* W2T = (bf16_t*)alloc((size_t)D_ * DF_ * 2);
  bf16_t* Qb  = (bf16_t*)alloc((size_t)M_ * D_ * 2);
  bf16_t* Kb  = (bf16_t*)alloc((size_t)M_ * D_ * 2);
  bf16_t* Vb  = (bf16_t*)alloc((size_t)M_ * D_ * 2);
  bf16_t* VTb = (bf16_t*)alloc((size_t)M_ * D_ * 2);
  bf16_t* att = (bf16_t*)alloc((size_t)M_ * D_ * 2);
  float*  proj = (float*)alloc((size_t)M_ * D_ * 4);
  float*  y1   = (float*)alloc((size_t)M_ * D_ * 4);
  bf16_t* y1b  = (bf16_t*)alloc((size_t)M_ * D_ * 2);
  bf16_t* hb   = (bf16_t*)alloc((size_t)M_ * DF_ * 2);
  float*  ffn  = (float*)alloc((size_t)M_ * D_ * 4);

  // 1) precision conversion / weight transposition
  {
    int n = M_ * D_;
    cvt_bf16_kernel<<<(n + 255) / 256, 256, 0, stream>>>(x, xb, n);
  }
  transpose_cvt_kernel<<<dim3(D_ / 32, D_ / 32), 256, 0, stream>>>(Wq, WqT, D_, D_);
  transpose_cvt_kernel<<<dim3(D_ / 32, D_ / 32), 256, 0, stream>>>(Wk, WkT, D_, D_);
  transpose_cvt_kernel<<<dim3(D_ / 32, D_ / 32), 256, 0, stream>>>(Wv, WvT, D_, D_);
  transpose_cvt_kernel<<<dim3(D_ / 32, D_ / 32), 256, 0, stream>>>(Wo, WoT, D_, D_);
  transpose_cvt_kernel<<<dim3(DF_ / 32, D_ / 32), 256, 0, stream>>>(W1, W1T, D_, DF_);
  transpose_cvt_kernel<<<dim3(D_ / 32, DF_ / 32), 256, 0, stream>>>(W2, W2T, DF_, D_);

  // 2) QKV projections (WMMA)
  gemm_bf16_kernel<0><<<dim3(M_ / 128, D_ / 128), 128, 0, stream>>>(xb, WqT, bq, Qb, M_, D_, D_);
  gemm_bf16_kernel<0><<<dim3(M_ / 128, D_ / 128), 128, 0, stream>>>(xb, WkT, bk, Kb, M_, D_, D_);
  gemm_bf16_kernel<0><<<dim3(M_ / 128, D_ / 128), 128, 0, stream>>>(xb, WvT, bv, Vb, M_, D_, D_);

  // 3) per-head V transpose, then flash attention (WMMA)
  transpose_v_kernel<<<dim3(N_ / 32, H_ * 2, B_), 256, 0, stream>>>(Vb, VTb);
  attention_kernel<<<dim3(N_ / 64, H_, B_), 128, 0, stream>>>(Qb, Kb, VTb, att);

  // 4) output projection + residual + LN1
  gemm_bf16_kernel<2><<<dim3(M_ / 128, D_ / 128), 128, 0, stream>>>(att, WoT, bo, proj, M_, D_, D_);
  resln_kernel<<<M_, 256, 0, stream>>>(proj, x, g1, be1, y1, y1b);

  // 5) FFN: GELU(y1 @ W1 + b1) @ W2 + b2, residual + LN2 -> out
  gemm_bf16_kernel<1><<<dim3(M_ / 128, DF_ / 128), 128, 0, stream>>>(y1b, W1T, b1, hb, M_, DF_, D_);
  gemm_bf16_kernel<2><<<dim3(M_ / 128, D_ / 128), 128, 0, stream>>>(hb, W2T, b2, ffn, M_, D_, DF_);
  resln_kernel<<<M_, 256, 0, stream>>>(ffn, y1, g2, be2, out, (bf16_t*)nullptr);
}
